// MultiLSTMConceptCorrector_27084063768956
// MI455X (gfx1250) — matrix operand order repaired and checked by reading, not compiled, not run on hardware
//
#include <hip/hip_runtime.h>
#include <hip/hip_bf16.h>

// ---------------- problem constants ----------------
#define B_   64
#define T_   256
#define C_   256
#define M_   8
#define H_   256
#define CC_  32          // concepts per cluster
#define KC   (CC_ + H_)  // 288: fused [x | h] K dimension
#define G4   (4 * H_)    // 1024 gate columns
#define XHPAD 296        // padded LDS row stride (bf16 elems) to break bank conflicts

// ---------------- vector types for WMMA ----------------
typedef __attribute__((ext_vector_type(16))) __bf16          v16bf;
typedef __attribute__((ext_vector_type(8)))  float           v8f;
typedef __attribute__((ext_vector_type(16))) unsigned short  v16s;
typedef __attribute__((ext_vector_type(8)))  unsigned short  v8s;

// fp32 -> bf16 bits, round-to-nearest-even (raw u16; no scalar bf16 math)
__device__ __forceinline__ unsigned short f2bf(float f) {
    unsigned u = __builtin_bit_cast(unsigned, f);
    unsigned r = u + 0x7FFFu + ((u >> 16) & 1u);
    return (unsigned short)(r >> 16);
}

__device__ __forceinline__ v16bf make16(v8s lo, v8s hi) {
    v16s t;
#pragma unroll
    for (int i = 0; i < 8; ++i) { t[i] = lo[i]; t[8 + i] = hi[i]; }
    return __builtin_bit_cast(v16bf, t);
}

__device__ __forceinline__ float fsig(float x)  { return 1.0f / (1.0f + __expf(-x)); }
__device__ __forceinline__ float ftanh(float x) { float e = __expf(2.0f * x); return (e - 1.0f) / (e + 1.0f); }

// ---------------- workspace layout (bytes) ----------------
#define WCAT_ELEMS ((size_t)M_ * G4 * KC)         // 2,359,296 bf16
#define WCAT_OFF   ((size_t)0)
#define BIAS_OFF   (WCAT_OFF + WCAT_ELEMS * 2)    // M*4H f32 combined bias
#define WFC_OFF    (BIAS_OFF + (size_t)M_ * G4 * 4)
#define WFC_ELEMS  ((size_t)M_ * CC_ * H_)        // 65,536 bf16
#define HS_OFF     (((WFC_OFF + WFC_ELEMS * 2) + 255) & ~(size_t)255)
#define HS_ELEMS   ((size_t)M_ * T_ * B_ * H_)    // 33,554,432 bf16 (64 MB)

// =====================================================================
// Kernel 0: one-shot packing of weights to bf16 + bias fusion.
// =====================================================================
__global__ void prep_kernel(const float* __restrict__ W_ih, const float* __restrict__ W_hh,
                            const float* __restrict__ b_ih, const float* __restrict__ b_hh,
                            const float* __restrict__ W_fc,
                            unsigned short* __restrict__ wcat, float* __restrict__ biasc,
                            unsigned short* __restrict__ wfc) {
    size_t i = (size_t)blockIdx.x * blockDim.x + threadIdx.x;
    if (i < WCAT_ELEMS) {
        size_t k = i % KC, row = i / KC;   // row = m*4H + col
        float v = (k < CC_) ? W_ih[row * CC_ + k] : W_hh[row * H_ + (k - CC_)];
        wcat[i] = f2bf(v);
        return;
    }
    i -= WCAT_ELEMS;
    if (i < (size_t)M_ * G4) { biasc[i] = b_ih[i] + b_hh[i]; return; }
    i -= (size_t)M_ * G4;
    if (i < WFC_ELEMS) wfc[i] = f2bf(W_fc[i]);
}

// =====================================================================
// Kernel 1: fused blend + input-projection + LSTM scan.
// grid = (M, B/16); block = 512 (16 waves). Wave w owns h-tile w for all
// 4 gates. Weight policy (no spills by construction):
//   - gates 0,1 (i,f): 18 v16bf tiles PINNED in VGPRs for the whole scan
//   - gates 2,3 (g,o): streamed from L2 each step; a laundered zero OFFSET
//     (not pointer!) defeats LICM while keeping global addrspace inference
//     so the loads stay global_load_b128 (FLAT would also tick DScnt).
// Per step per wave: 9 K-chunks x 4 gates = 36 v_wmma_f32_16x16x32_bf16.
// =====================================================================
__global__ __launch_bounds__(512) void lstm_scan_kernel(
    const float* __restrict__ inputs, const float* __restrict__ aic,
    const float* __restrict__ orig,   const float* __restrict__ h0,
    const float* __restrict__ c0,
    const unsigned short* __restrict__ wcat, const float* __restrict__ biasc,
    unsigned short* __restrict__ hs, float* __restrict__ hT, float* __restrict__ cT) {

    __shared__ __align__(16) unsigned short xh[16][XHPAD];  // [batch row][ x(32) | h(256) ]

    const int m    = blockIdx.x;
    const int b0   = blockIdx.y * 16;
    const int tid  = threadIdx.x;
    const int lane = tid & 31;
    const int wave = tid >> 5;          // 0..15 -> h tile
    const int half = (lane >= 16);
    const int ln   = lane & 15;

    const unsigned short* wm    = wcat + (size_t)m * G4 * KC;
    const float*          biasm = biasc + (size_t)m * G4;

    // ---- pin gates 0,1 weights in VGPRs (18 x v16bf = 144 VGPRs/lane) ----
    v16bf wreg[18];
#pragma unroll
    for (int g = 0; g < 2; ++g) {
        int col = g * H_ + wave * 16 + ln;   // B operand: n = ln, 16 contiguous K per half
#pragma unroll
        for (int kk = 0; kk < 9; ++kk) {
            const unsigned short* bp = wm + (size_t)col * KC + kk * 32 + 16 * half;
            wreg[g * 9 + kk] = __builtin_bit_cast(v16bf, *(const v16s*)bp);
        }
    }

    // ---- hoist the 4 per-lane bias values (loop-invariant) ----
    float bias_g[4];
#pragma unroll
    for (int g = 0; g < 4; ++g) bias_g[g] = biasm[g * H_ + wave * 16 + ln];

    // h0 -> LDS cols [32, 288)
    for (int i = tid; i < 16 * H_; i += 512) {
        int r = i >> 8, hc = i & 255;
        xh[r][CC_ + hc] = f2bf(h0[((size_t)m * B_ + b0 + r) * H_ + hc]);
    }
    // c0 -> registers in WMMA C layout: lane ln holds col, VGPR v holds row v+8*half
    v8f cs;
#pragma unroll
    for (int v = 0; v < 8; ++v) {
        int r = v + 8 * half;
        cs[v] = c0[((size_t)m * B_ + b0 + r) * H_ + wave * 16 + ln];
    }
    __syncthreads();

    const unsigned short* arow = &xh[ln][0];

    for (int t = 0; t < T_; ++t) {
        // laundered zero offset: blocks LICM of the streamed weight loads but
        // keeps pointer provenance (=> global_load, not flat_load)
        int zoff = 0;
        asm volatile("" : "+s"(zoff));
        const unsigned short* wmt = wm + zoff;

        // blend x_t into LDS cols [0,32): 512 threads, 1 elem each (128B/wave coalesced)
        {
            int r = tid >> 5, cc = tid & 31;
            size_t gi = (((size_t)(b0 + r)) * T_ + t) * C_ + m * CC_ + cc;
            float a  = aic[gi];
            float xv = a * inputs[gi] + (1.0f - a) * orig[gi];
            xh[r][cc] = f2bf(xv);
        }
        __syncthreads();

        // g = [x|h] @ Wcat^T + bias  (16x1024 per WG, K=288)
        v8f acc[4];
#pragma unroll
        for (int g = 0; g < 4; ++g) {
#pragma unroll
            for (int v = 0; v < 8; ++v) acc[g][v] = bias_g[g];
        }
#pragma unroll
        for (int kk = 0; kk < 9; ++kk) {
            int kb = kk * 32;
            // A lane layout: 16 contiguous K per half (matches B)
            v8s alo = *(const v8s*)(arow + kb + 16 * half);
            v8s ahi = *(const v8s*)(arow + kb + 16 * half + 8);
            v16bf av = make16(alo, ahi);

            // gates 0,1: register-resident weights
            acc[0] = __builtin_amdgcn_wmma_f32_16x16x32_bf16(
                false, av, false, wreg[kk], (short)0, acc[0], false, false);
            acc[1] = __builtin_amdgcn_wmma_f32_16x16x32_bf16(
                false, av, false, wreg[9 + kk], (short)0, acc[1], false, false);

            // gates 2,3: streamed from L2 (weights stay hot: 4.7 MB total vs 192 MB L2)
#pragma unroll
            for (int g = 2; g < 4; ++g) {
                int col = g * H_ + wave * 16 + ln;
                const unsigned short* bp = wmt + (size_t)col * KC + kb + 16 * half;
                v16bf bv = __builtin_bit_cast(v16bf, *(const v16s*)bp);
                acc[g] = __builtin_amdgcn_wmma_f32_16x16x32_bf16(
                    false, av, false, bv, (short)0, acc[g], false, false);
            }
        }
        __syncthreads();  // all waves done reading xh

        // elementwise LSTM cell (gate order i,f,g,o), write h back
#pragma unroll
        for (int v = 0; v < 8; ++v) {
            int r = v + 8 * half;
            float ig = fsig(acc[0][v]);
            float fg = fsig(acc[1][v]);
            float gg = ftanh(acc[2][v]);
            float og = fsig(acc[3][v]);
            float cn = fg * cs[v] + ig * gg;
            cs[v] = cn;
            float hn = og * ftanh(cn);
            unsigned short hb = f2bf(hn);
            xh[r][CC_ + wave * 16 + ln] = hb;
            hs[(((size_t)m * T_ + t) * B_ + b0 + r) * H_ + wave * 16 + ln] = hb;
            if (t == T_ - 1) {
                size_t oi = ((size_t)m * B_ + b0 + r) * H_ + wave * 16 + ln;
                hT[oi] = hn;
                cT[oi] = cn;
            }
        }
        __syncthreads();
    }
}

// =====================================================================
// Kernel 2: fc = sigmoid(hs @ W_fc^T + b_fc); out = a*orig + (1-a)*fc.
// Fully parallel: rows = T*B = 16384, cols = 32. grid=(M,128), block=256.
// =====================================================================
__global__ __launch_bounds__(256) void fc_blend_kernel(
    const unsigned short* __restrict__ hs, const unsigned short* __restrict__ wfc,
    const float* __restrict__ bfc, const float* __restrict__ aic,
    const float* __restrict__ orig, float* __restrict__ out) {

    const int m    = blockIdx.x;
    const int tid  = threadIdx.x;
    const int lane = tid & 31;
    const int wave = tid >> 5;                 // 8 waves
    const int half = (lane >= 16);
    const int ln   = lane & 15;
    const int rt   = blockIdx.y * 8 + wave;    // row tile over flattened (t,b)

    const unsigned short* hsm = hs  + (size_t)m * T_ * B_ * H_;
    const unsigned short* wm  = wfc + (size_t)m * CC_ * H_;

    v8f acc[2];
#pragma unroll
    for (int ct = 0; ct < 2; ++ct) {
        float bv = bfc[m * CC_ + ct * 16 + ln];
#pragma unroll
        for (int v = 0; v < 8; ++v) acc[ct][v] = bv;
    }

    const unsigned short* ap = hsm + ((size_t)rt * 16 + ln) * H_;
#pragma unroll
    for (int kk = 0; kk < 8; ++kk) {
        int kb = kk * 32;
        v8s alo = *(const v8s*)(ap + kb + 16 * half);
        v8s ahi = *(const v8s*)(ap + kb + 16 * half + 8);
        v16bf av = make16(alo, ahi);
#pragma unroll
        for (int ct = 0; ct < 2; ++ct) {
            const unsigned short* bp = wm + (size_t)(ct * 16 + ln) * H_ + kb + 16 * half;
            v16bf bv = __builtin_bit_cast(v16bf, *(const v16s*)bp);
            acc[ct] = __builtin_amdgcn_wmma_f32_16x16x32_bf16(
                false, av, false, bv, (short)0, acc[ct], false, false);
        }
    }

#pragma unroll
    for (int ct = 0; ct < 2; ++ct) {
#pragma unroll
        for (int v = 0; v < 8; ++v) {
            int r  = rt * 16 + v + 8 * half;    // flattened (t,b): t = r/64, b = r%64
            int tt = r >> 6, bb = r & 63;
            size_t gi = (((size_t)bb) * T_ + tt) * C_ + m * CC_ + ct * 16 + ln;
            float a  = aic[gi];
            float fc = fsig(acc[ct][v]);
            out[gi]  = a * orig[gi] + (1.0f - a) * fc;
        }
    }
}

// =====================================================================
extern "C" void kernel_launch(void* const* d_in, const int* in_sizes, int n_in,
                              void* d_out, int out_size, void* d_ws, size_t ws_size,
                              hipStream_t stream) {
    (void)in_sizes; (void)n_in; (void)out_size; (void)ws_size;

    const float* inputs = (const float*)d_in[0];
    const float* aic    = (const float*)d_in[1];
    const float* orig   = (const float*)d_in[2];
    const float* h0     = (const float*)d_in[3];
    const float* c0     = (const float*)d_in[4];
    const float* W_ih   = (const float*)d_in[5];
    const float* W_hh   = (const float*)d_in[6];
    const float* b_ih   = (const float*)d_in[7];
    const float* b_hh   = (const float*)d_in[8];
    const float* W_fc   = (const float*)d_in[9];
    const float* b_fc   = (const float*)d_in[10];

    float* out = (float*)d_out;
    float* hT  = out + (size_t)B_ * T_ * C_;
    float* cT  = hT + (size_t)M_ * B_ * H_;

    char* ws = (char*)d_ws;
    unsigned short* wcat  = (unsigned short*)(ws + WCAT_OFF);
    float*          biasc = (float*)(ws + BIAS_OFF);
    unsigned short* wfc   = (unsigned short*)(ws + WFC_OFF);
    unsigned short* hs    = (unsigned short*)(ws + HS_OFF);

    // 0) pack weights to bf16, fuse biases
    size_t nprep = WCAT_ELEMS + (size_t)M_ * G4 + WFC_ELEMS;
    prep_kernel<<<(unsigned)((nprep + 255) / 256), 256, 0, stream>>>(
        W_ih, W_hh, b_ih, b_hh, W_fc, wcat, biasc, wfc);

    // 1) fused blend + projection + LSTM scan (serial in T, parallel over M x batch tiles)
    dim3 g1(M_, B_ / 16);
    lstm_scan_kernel<<<g1, 512, 0, stream>>>(
        inputs, aic, orig, h0, c0, wcat, biasc, hs, hT, cT);

    // 2) FC + sigmoid + output blend (fully parallel)
    dim3 g2(M_, (T_ * B_) / (16 * 8));
    fc_blend_kernel<<<g2, 256, 0, stream>>>(hs, wfc, b_fc, aic, orig, out);
}
